// KNNClassifier_15908558865114
// MI455X (gfx1250) — matrix operand order repaired and checked by reading, not compiled, not run on hardware
//
#include <hip/hip_runtime.h>
#include <stddef.h>

typedef __attribute__((ext_vector_type(16))) __bf16 v16bf;
typedef __attribute__((ext_vector_type(8)))  float  v8f;

constexpr int D      = 512;
constexpr int BQ     = 2048;      // batch of queries
constexpr int N      = 100000;    // train points
constexpr int NC     = 1000;      // classes
constexpr int KNN    = 16;
constexpr int KSTEPS = D / 32;    // 16 wmma k-steps
constexpr int NTILES = N / 16;    // 6250
constexpr int MTILES = BQ / 16;   // 128
constexpr int NSPLIT = 25;        // column partitions
constexpr int TPS    = NTILES / NSPLIT; // 250 tiles per split
constexpr float FINF = 3.0e38f;

// ---------------------------------------------------------------------------
// Pack Xtrain (row-major f32 [N,D]) into bf16 B-matrix WMMA fragments.
// Fragment f = nt*KSTEPS + ks covers K rows [ks*32, ks*32+32) x N cols
// [nt*16, nt*16+16).  Contiguous k order, same as A (32x16 16-bit B):
//   lane n = lane&15, kHalf = lane>>4, element e: K = ks*32 + 16*kHalf + e.
// Stored as [frag][lane][e] contiguous -> one 32B load per lane in the GEMM.
// ---------------------------------------------------------------------------
__global__ void pack_b_bf16(const float* __restrict__ Xtrain,
                            __bf16* __restrict__ Bpack) {
  int gw   = blockIdx.x * 8 + (threadIdx.x >> 5);   // fragment id, 100000 total
  int lane = threadIdx.x & 31;
  int nt = gw >> 4, ks = gw & 15;
  int n     = nt * 16 + (lane & 15);
  int kbase = ks * 32 + ((lane >> 4) << 4);
  const float4* src = (const float4*)(Xtrain + (size_t)n * D + kbase);
  float4 f0 = src[0], f1 = src[1], f2 = src[2], f3 = src[3];
  v16bf frag;
  frag[0]=(__bf16)f0.x; frag[1]=(__bf16)f0.y; frag[2]=(__bf16)f0.z; frag[3]=(__bf16)f0.w;
  frag[4]=(__bf16)f1.x; frag[5]=(__bf16)f1.y; frag[6]=(__bf16)f1.z; frag[7]=(__bf16)f1.w;
  frag[8]=(__bf16)f2.x; frag[9]=(__bf16)f2.y; frag[10]=(__bf16)f2.z; frag[11]=(__bf16)f2.w;
  frag[12]=(__bf16)f3.x; frag[13]=(__bf16)f3.y; frag[14]=(__bf16)f3.z; frag[15]=(__bf16)f3.w;
  *(v16bf*)(Bpack + (size_t)gw * 512 + lane * 16) = frag;
}

// ---------------------------------------------------------------------------
// Pack x (f32 [B,D]) into bf16 A-matrix WMMA fragments.
// Contiguous k order, same as B (16x32 16-bit A): lane m = lane&15, kHalf = lane>>4,
//   element e: K = ks*32 + 16*kHalf + e
// ---------------------------------------------------------------------------
__global__ void pack_a_bf16(const float* __restrict__ x,
                            __bf16* __restrict__ Apack) {
  int gw   = blockIdx.x * 8 + (threadIdx.x >> 5);   // fragment id, 2048 total
  int lane = threadIdx.x & 31;
  int mt = gw >> 4, ks = gw & 15;
  int m  = mt * 16 + (lane & 15);
  int kh = lane >> 4;
  int k0 = ks * 32;
  const float* xr = x + (size_t)m * D;
  float4 a0 = *(const float4*)(xr + k0 + 16 * kh);
  float4 a1 = *(const float4*)(xr + k0 + 16 * kh + 4);
  float4 b0 = *(const float4*)(xr + k0 + 16 * kh + 8);
  float4 b1 = *(const float4*)(xr + k0 + 16 * kh + 12);
  v16bf frag;
  frag[0]=(__bf16)a0.x; frag[1]=(__bf16)a0.y; frag[2]=(__bf16)a0.z; frag[3]=(__bf16)a0.w;
  frag[4]=(__bf16)a1.x; frag[5]=(__bf16)a1.y; frag[6]=(__bf16)a1.z; frag[7]=(__bf16)a1.w;
  frag[8]=(__bf16)b0.x; frag[9]=(__bf16)b0.y; frag[10]=(__bf16)b0.z; frag[11]=(__bf16)b0.w;
  frag[12]=(__bf16)b1.x; frag[13]=(__bf16)b1.y; frag[14]=(__bf16)b1.z; frag[15]=(__bf16)b1.w;
  *(v16bf*)(Apack + (size_t)gw * 512 + lane * 16) = frag;
}

// One wave per row: squared L2 norm of a D=512 row (coalesced).
__global__ void row_norm2(const float* __restrict__ X, float* __restrict__ out,
                          int rows) {
  int r = blockIdx.x * 8 + (threadIdx.x >> 5);
  if (r >= rows) return;
  int lane = threadIdx.x & 31;
  const float* p = X + (size_t)r * D;
  float s = 0.0f;
  #pragma unroll
  for (int i = 0; i < D / 32; i++) { float v = p[lane + 32 * i]; s = fmaf(v, v, s); }
  #pragma unroll
  for (int off = 16; off > 0; off >>= 1) s += __shfl_xor(s, off, 32);
  if (lane == 0) out[r] = s;
}

__global__ void zero_f32(float* __restrict__ p, int n) {
  int i = blockIdx.x * blockDim.x + threadIdx.x;
  if (i < n) p[i] = 0.0f;
}

// ---------------------------------------------------------------------------
// Fused WMMA distance GEMM + per-row top-16 over an N-slice.
// Block = (mt, split): 16 query rows x 4000 train columns, 8 waves.
// key = ||t||^2 - 2*dot  (||x||^2 added in merge; constant per row -> order-safe)
// K-reduction uses TWO independent accumulator chains (even/odd k-steps) to
// halve the WMMA D->C RAW dependency depth; summed at the end.
// ---------------------------------------------------------------------------
__global__ __launch_bounds__(256, 1) void knn_wmma_topk(
    const __bf16* __restrict__ Apack, const __bf16* __restrict__ Bpack,
    const float* __restrict__ tnorm2,
    float* __restrict__ partD, int* __restrict__ partI) {
  __shared__ __bf16 ldsA[16 * 512];        // 16 A fragments for this mt (16 KB)
  __shared__ float  ldsD[8][16][16];       // per-wave 16x16 key tile
  __shared__ float  ldsMD[16][128];        // merge buffer: 8 waves x 16 cand
  __shared__ int    ldsMI[16][128];

  int mt = blockIdx.x, split = blockIdx.y;
  int tid = threadIdx.x, w = tid >> 5, lane = tid & 31;
  int laneN = lane & 15, laneH = lane >> 4;

  // Cooperative straight copy of the packed A tile (layouts match).
  {
    const uint4* srcA = (const uint4*)(Apack + (size_t)mt * 16 * 512);
    uint4* dstA = (uint4*)ldsA;
    for (int i = tid; i < 1024; i += 256) dstA[i] = srcA[i];
  }
  __syncthreads();

  float hd[KNN]; int hix[KNN]; float hmax = FINF;
  #pragma unroll
  for (int j = 0; j < KNN; j++) { hd[j] = FINF; hix[j] = 0; }

  for (int t = w; t < TPS; t += 8) {
    int nt = split * TPS + t;
    const __bf16* bptr = Bpack + (size_t)nt * 16 * 512 + lane * 16;
    __builtin_prefetch(bptr + (size_t)8 * 16 * 512, 0, 3);  // next tile, near-cache
    v8f acc0 = {};
    v8f acc1 = {};
    #pragma unroll
    for (int ks = 0; ks < KSTEPS; ks += 2) {
      v16bf a0 = *(const v16bf*)(ldsA + ks * 512 + lane * 16);
      v16bf b0 = *(const v16bf*)(bptr + ks * 512);
      acc0 = __builtin_amdgcn_wmma_f32_16x16x32_bf16(
          false, a0, false, b0, (short)0, acc0, false, false);
      v16bf a1 = *(const v16bf*)(ldsA + (ks + 1) * 512 + lane * 16);
      v16bf b1 = *(const v16bf*)(bptr + (ks + 1) * 512);
      acc1 = __builtin_amdgcn_wmma_f32_16x16x32_bf16(
          false, a1, false, b1, (short)0, acc1, false, false);
    }
    v8f acc = acc0 + acc1;
    float tn = tnorm2[nt * 16 + laneN];
    // C layout: VGPR r holds M = r + 8*laneH, N = laneN
    #pragma unroll
    for (int r = 0; r < 8; r++)
      ldsD[w][r + (laneH << 3)][laneN] = __builtin_fmaf(-2.0f, acc[r], tn);

    if (lane < 16) {  // lane owns query row `lane`; scan 16 columns
      for (int j = 0; j < 16; j++) {
        float d = ldsD[w][lane][j];
        int idx = nt * 16 + j;
        if (d < hmax) {
          bool rep = true;
          #pragma unroll
          for (int q = 0; q < KNN; q++)
            if (rep && hd[q] == hmax) { hd[q] = d; hix[q] = idx; rep = false; }
          float m = hd[0];
          #pragma unroll
          for (int q = 1; q < KNN; q++) m = fmaxf(m, hd[q]);
          hmax = m;
        }
      }
    }
  }

  if (lane < 16) {
    #pragma unroll
    for (int j = 0; j < KNN; j++) {
      ldsMD[lane][w * 16 + j] = hd[j];
      ldsMI[lane][w * 16 + j] = hix[j];
    }
  }
  __syncthreads();

  if (w == 0 && lane < 16) {  // merge 8 wave-heaps -> sorted top-16 per row
    int row = mt * 16 + lane;
    float* pD = partD + ((size_t)row * NSPLIT + split) * KNN;
    int*   pI = partI + ((size_t)row * NSPLIT + split) * KNN;
    for (int k = 0; k < KNN; k++) {
      float bv = FINF; int bj = 0;
      for (int j = 0; j < 128; j++) {
        float v = ldsMD[lane][j];
        if (v < bv) { bv = v; bj = j; }
      }
      pD[k] = bv; pI[k] = ldsMI[lane][bj];
      ldsMD[lane][bj] = FINF;
    }
  }
}

// ---------------------------------------------------------------------------
// Final 25-way sorted merge per row + inverse-distance weighted label vote.
// One thread owns one row -> no atomics needed.
// ---------------------------------------------------------------------------
__global__ void knn_merge_vote(const float* __restrict__ partD,
                               const int* __restrict__ partI,
                               const float* __restrict__ xnorm2,
                               const int* __restrict__ ytrain,
                               float* __restrict__ logits) {
  int row = blockIdx.x * blockDim.x + threadIdx.x;
  if (row >= BQ) return;
  int ptr[NSPLIT];
  for (int l = 0; l < NSPLIT; l++) ptr[l] = 0;
  float xn = xnorm2[row];
  const float* pd = partD + (size_t)row * NSPLIT * KNN;
  const int*   pi = partI + (size_t)row * NSPLIT * KNN;
  for (int k = 0; k < KNN; k++) {
    float bv = FINF; int bl = 0;
    for (int l = 0; l < NSPLIT; l++) {
      int p = ptr[l];
      if (p < KNN) {
        float v = pd[l * KNN + p];
        if (v < bv) { bv = v; bl = l; }
      }
    }
    int idx = pi[bl * KNN + ptr[bl]];
    ptr[bl]++;
    float dist = bv + xn;
    float wgt = 1.0f / fmaxf(dist, 1e-6f);
    int label = ytrain[idx];
    logits[(size_t)row * NC + label] += wgt;
  }
}

// ---------------------------------------------------------------------------
extern "C" void kernel_launch(void* const* d_in, const int* in_sizes, int n_in,
                              void* d_out, int out_size, void* d_ws, size_t ws_size,
                              hipStream_t stream) {
  const float* x      = (const float*)d_in[0];   // [2048,512]
  const float* Xtrain = (const float*)d_in[1];   // [100000,512]
  const int*   ytrain = (const int*)d_in[2];     // [100000]
  float* logits = (float*)d_out;                 // [2048,1000]
  (void)in_sizes; (void)n_in; (void)out_size; (void)ws_size;

  char* ws = (char*)d_ws;
  size_t off = 0;
  auto alloc = [&](size_t bytes) { char* p = ws + off; off += (bytes + 255) & ~size_t(255); return p; };
  __bf16* Bpack  = (__bf16*)alloc((size_t)N * D * sizeof(__bf16));   // 102.4 MB
  __bf16* Apack  = (__bf16*)alloc((size_t)BQ * D * sizeof(__bf16));  //   2.1 MB
  float*  tnorm2 = (float*)alloc((size_t)N * sizeof(float));
  float*  xnorm2 = (float*)alloc((size_t)BQ * sizeof(float));
  float*  partD  = (float*)alloc((size_t)BQ * NSPLIT * KNN * sizeof(float));
  int*    partI  = (int*)alloc((size_t)BQ * NSPLIT * KNN * sizeof(int));

  pack_b_bf16<<<NTILES * KSTEPS / 8, 256, 0, stream>>>(Xtrain, Bpack);   // 12500 blocks
  pack_a_bf16<<<MTILES * KSTEPS / 8, 256, 0, stream>>>(x, Apack);        // 256 blocks
  row_norm2<<<(N + 7) / 8, 256, 0, stream>>>(Xtrain, tnorm2, N);
  row_norm2<<<(BQ + 7) / 8, 256, 0, stream>>>(x, xnorm2, BQ);
  zero_f32<<<(BQ * NC + 255) / 256, 256, 0, stream>>>(logits, BQ * NC);

  dim3 grid(MTILES, NSPLIT);
  knn_wmma_topk<<<grid, 256, 0, stream>>>(Apack, Bpack, tnorm2, partD, partI);

  knn_merge_vote<<<(BQ + 255) / 256, 256, 0, stream>>>(partD, partI, xnorm2, ytrain, logits);
}
